// GraphCastEncoder_40321152975369
// MI455X (gfx1250) — matrix operand order repaired and checked by reading, not compiled, not run on hardware
//
#include <hip/hip_runtime.h>
#include <cstdint>
#include <cstddef>

#define HD      512
#define E_EDGES 131072
#define NMESH   40962
#define NHALO   4096
#define NGRID   131072
#define ROWS    32          // rows per block (2 row-tiles of 16)

typedef __attribute__((ext_vector_type(16))) __bf16 v16bf;
typedef __attribute__((ext_vector_type(8)))  __bf16 v8bf;
typedef __attribute__((ext_vector_type(8)))  float  v8f;

__device__ __forceinline__ __bf16 f2bf(float f) {
    union { float f; unsigned u; } in; in.f = f;
    unsigned r = (in.u + 0x7FFFu + ((in.u >> 16) & 1u)) >> 16;  // RNE
    union { unsigned short s; __bf16 b; } out; out.s = (unsigned short)r;
    return out.b;
}

__device__ __forceinline__ v16bf cat16(v8bf lo, v8bf hi) {
    return __builtin_shufflevector(lo, hi, 0,1,2,3,4,5,6,7,8,9,10,11,12,13,14,15);
}

// ---------------------------------------------------------------------------
// Core: 32 x DIN bf16 tile staged at smem base -> compute
//   LN-stats( silu(A@W1+b1) @ W2 + b2 )
// leaving fp32 32x512 tile at smem base (OutT), mu in rowstat[0..31],
// rstd in rowstat[32..63].  smem reused (A -> H -> Out) across barriers.
// 8 waves: wave w -> row-tile (w>>2), columns (w&3)*128 .. +127 (8 col-tiles).
// ---------------------------------------------------------------------------
template<int DIN>
__device__ __forceinline__ void mlp_tile_core(
    unsigned char* smem, float* red, float* rowstat,
    const __bf16* __restrict__ w1t, const float* __restrict__ b1,
    const __bf16* __restrict__ w2t, const float* __restrict__ b2)
{
    const int tid     = threadIdx.x;
    const int lane    = tid & 31;
    const int ln      = lane & 15;          // A row / B col within 16x16 tile
    const int hi      = lane >> 4;          // lane half selects K sub-range
    const int wave    = tid >> 5;
    const int rt      = wave >> 2;          // row tile (0 or 1)
    const int colBase = (wave & 3) * 128;   // wave's 128-column slab

    __bf16* Atile = (__bf16*)smem;          // 32 x DIN  (phase 1)
    __bf16* Htile = (__bf16*)smem;          // 32 x 512  (phase 2, aliases A)
    float*  OutT  = (float*) smem;          // 32 x 512  (phase 3, aliases H)

    const v8f vzero = {0.f,0.f,0.f,0.f,0.f,0.f,0.f,0.f};
    const int arow = rt * 16 + ln;

    // -------- GEMM1: H = A(32xDIN) @ W1(DINx512) --------
    v8f acc[8];
    #pragma unroll
    for (int t = 0; t < 8; ++t) acc[t] = vzero;

    for (int kb = 0; kb < DIN; kb += 32) {
        const __bf16* ap = Atile + arow * DIN + kb + hi * 8;
        v16bf a = cat16(*(const v8bf*)ap, *(const v8bf*)(ap + 16));
        #pragma unroll
        for (int t = 0; t < 8; ++t) {
            const __bf16* bp = w1t + (size_t)(colBase + t*16 + ln) * DIN + kb + hi * 8;
            v16bf b = cat16(*(const v8bf*)bp, *(const v8bf*)(bp + 16));
            acc[t] = __builtin_amdgcn_wmma_f32_16x16x32_bf16(
                false, a, false, b, (short)0, acc[t], false, false);
        }
    }
    __syncthreads();   // Atile dead -> safe to overwrite with Htile

    #pragma unroll
    for (int t = 0; t < 8; ++t) {
        const int col = colBase + t*16 + ln;
        const float bias = b1[col];
        #pragma unroll
        for (int j = 0; j < 8; ++j) {
            float v = acc[t][j] + bias;
            v = v / (1.0f + __expf(-v));              // SiLU
            Htile[(rt*16 + hi*8 + j) * HD + col] = f2bf(v);
        }
    }
    __syncthreads();

    // -------- GEMM2: Out = H(32x512) @ W2(512x512) --------
    v8f acc2[8];
    #pragma unroll
    for (int t = 0; t < 8; ++t) acc2[t] = vzero;

    for (int kb = 0; kb < HD; kb += 32) {
        const __bf16* ap = Htile + arow * HD + kb + hi * 8;
        v16bf a = cat16(*(const v8bf*)ap, *(const v8bf*)(ap + 16));
        #pragma unroll
        for (int t = 0; t < 8; ++t) {
            const __bf16* bp = w2t + (size_t)(colBase + t*16 + ln) * HD + kb + hi * 8;
            v16bf b = cat16(*(const v8bf*)bp, *(const v8bf*)(bp + 16));
            acc2[t] = __builtin_amdgcn_wmma_f32_16x16x32_bf16(
                false, a, false, b, (short)0, acc2[t], false, false);
        }
    }
    __syncthreads();   // Htile dead -> safe to overwrite with OutT

    #pragma unroll
    for (int t = 0; t < 8; ++t) {
        const int col = colBase + t*16 + ln;
        const float bias = b2[col];
        #pragma unroll
        for (int j = 0; j < 8; ++j)
            OutT[(rt*16 + hi*8 + j) * HD + col] = acc2[t][j] + bias;
    }
    __syncthreads();

    // -------- LayerNorm stats per row (8 threads / row, 32 rows) --------
    const int row = tid >> 3;
    const int sub = tid & 7;

    float s = 0.f;
    for (int c = sub; c < HD; c += 8) s += OutT[row * HD + c];
    red[tid] = s;
    __syncthreads();
    if (sub == 0) {
        float tot = 0.f;
        #pragma unroll
        for (int i = 0; i < 8; ++i) tot += red[row * 8 + i];
        rowstat[row] = tot * (1.0f / HD);
    }
    __syncthreads();
    const float mu = rowstat[row];
    float v = 0.f;
    for (int c = sub; c < HD; c += 8) {
        float d = OutT[row * HD + c] - mu;
        v += d * d;
    }
    red[tid] = v;
    __syncthreads();
    if (sub == 0) {
        float tot = 0.f;
        #pragma unroll
        for (int i = 0; i < 8; ++i) tot += red[row * 8 + i];
        rowstat[32 + row] = rsqrtf(tot * (1.0f / HD) + 1e-5f);
    }
    __syncthreads();
}

// ---------------------------------------------------------------------------
// Edge MLP: gather endpoints, MLP+LN, +efeat residual, atomic segment-sum
// ---------------------------------------------------------------------------
__global__ void __launch_bounds__(256)
edge_mlp_kernel(const float* __restrict__ efeat,
                const float* __restrict__ mesh,
                const float* __restrict__ halo,
                const long long* __restrict__ eidx,
                const __bf16* __restrict__ w1t, const float* __restrict__ b1,
                const __bf16* __restrict__ w2t, const float* __restrict__ b2,
                const float* __restrict__ g,   const float* __restrict__ beta,
                float* __restrict__ agg)
{
    __shared__ __align__(16) unsigned char smem[ROWS * 1536 * 2];  // 96 KB
    __shared__ float red[256];
    __shared__ float rowstat[64];
    __shared__ int   dstIdx[ROWS];

    const int tid = threadIdx.x;
    const int row = tid >> 3;
    const int sub = tid & 7;
    const int e   = blockIdx.x * ROWS + row;

    const long long d = eidx[2 * (long long)e + 0];
    const long long s = eidx[2 * (long long)e + 1];
    if (sub == 0) dstIdx[row] = (int)d;

    const float* srcp = (s < NMESH) ? (mesh + (size_t)s * HD)
                                    : (halo + (size_t)(s - NMESH) * HD);
    const float* dstp = mesh + (size_t)d * HD;
    const float* ep   = efeat + (size_t)e * HD;

    __bf16* Atile = (__bf16*)smem;
    for (int c = sub; c < HD; c += 8) {
        Atile[row * 1536 + c]        = f2bf(ep[c]);
        Atile[row * 1536 + 512 + c]  = f2bf(srcp[c]);
        Atile[row * 1536 + 1024 + c] = f2bf(dstp[c]);
    }
    __syncthreads();

    mlp_tile_core<1536>(smem, red, rowstat, w1t, b1, w2t, b2);

    float* OutT = (float*)smem;
    const float mu   = rowstat[row];
    const float rstd = rowstat[32 + row];
    const int   dn   = dstIdx[row];
    for (int c = sub; c < HD; c += 8) {
        float val = (OutT[row * HD + c] - mu) * rstd * g[c] + beta[c] + ep[c];
        atomicAdd(agg + (size_t)dn * HD + c, val);   // segment_sum over dst
    }
}

// ---------------------------------------------------------------------------
// Node MLP: [mesh | agg] -> MLP+LN, mesh_out = mesh + n_feats
// ---------------------------------------------------------------------------
__global__ void __launch_bounds__(256)
node_mlp_kernel(const float* __restrict__ mesh,
                const float* __restrict__ agg,
                const __bf16* __restrict__ w1t, const float* __restrict__ b1,
                const __bf16* __restrict__ w2t, const float* __restrict__ b2,
                const float* __restrict__ g,   const float* __restrict__ beta,
                float* __restrict__ mesh_out)
{
    __shared__ __align__(16) unsigned char smem[ROWS * 512 * 4];   // 64 KB (holds 32x1024 bf16 / 32x512 f32)
    __shared__ float red[256];
    __shared__ float rowstat[64];

    const int tid  = threadIdx.x;
    const int row  = tid >> 3;
    const int sub  = tid & 7;
    const int node = blockIdx.x * ROWS + row;
    const int nc   = (node < NMESH) ? node : (NMESH - 1);

    __bf16* Atile = (__bf16*)smem;
    for (int c = sub; c < HD; c += 8) {
        Atile[row * 1024 + c]       = f2bf(mesh[(size_t)nc * HD + c]);
        Atile[row * 1024 + 512 + c] = f2bf(agg[(size_t)nc * HD + c]);
    }
    __syncthreads();

    mlp_tile_core<1024>(smem, red, rowstat, w1t, b1, w2t, b2);

    if (node < NMESH) {
        float* OutT = (float*)smem;
        const float mu   = rowstat[row];
        const float rstd = rowstat[32 + row];
        for (int c = sub; c < HD; c += 8) {
            float val = (OutT[row * HD + c] - mu) * rstd * g[c] + beta[c];
            mesh_out[(size_t)node * HD + c] = mesh[(size_t)node * HD + c] + val;
        }
    }
}

// ---------------------------------------------------------------------------
// Grid MLP: grid_out = grid + LN(MLP(grid))
// ---------------------------------------------------------------------------
__global__ void __launch_bounds__(256)
grid_mlp_kernel(const float* __restrict__ grid,
                const __bf16* __restrict__ w1t, const float* __restrict__ b1,
                const __bf16* __restrict__ w2t, const float* __restrict__ b2,
                const float* __restrict__ g,   const float* __restrict__ beta,
                float* __restrict__ grid_out)
{
    __shared__ __align__(16) unsigned char smem[ROWS * 512 * 4];   // 64 KB
    __shared__ float red[256];
    __shared__ float rowstat[64];

    const int tid = threadIdx.x;
    const int row = tid >> 3;
    const int sub = tid & 7;
    const int n   = blockIdx.x * ROWS + row;

    __bf16* Atile = (__bf16*)smem;
    for (int c = sub; c < HD; c += 8)
        Atile[row * 512 + c] = f2bf(grid[(size_t)n * HD + c]);
    __syncthreads();

    mlp_tile_core<512>(smem, red, rowstat, w1t, b1, w2t, b2);

    float* OutT = (float*)smem;
    const float mu   = rowstat[row];
    const float rstd = rowstat[32 + row];
    for (int c = sub; c < HD; c += 8) {
        float val = (OutT[row * HD + c] - mu) * rstd * g[c] + beta[c];
        grid_out[(size_t)n * HD + c] = grid[(size_t)n * HD + c] + val;
    }
}

// ---------------------------------------------------------------------------
// Helpers
// ---------------------------------------------------------------------------
__global__ void zero_kernel(float* __restrict__ p, int n) {
    int i = blockIdx.x * 256 + threadIdx.x;
    if (i < n) p[i] = 0.0f;
}

__global__ void convert_transpose_kernel(const float* __restrict__ w,
                                         __bf16* __restrict__ wt,
                                         int din, int dout) {
    int i = blockIdx.x * 256 + threadIdx.x;
    if (i >= din * dout) return;
    int n = i / din;
    int k = i - n * din;
    wt[i] = f2bf(w[(size_t)k * dout + n]);   // wt[n*din + k] = w[k, n]
}

// ---------------------------------------------------------------------------
extern "C" void kernel_launch(void* const* d_in, const int* in_sizes, int n_in,
                              void* d_out, int out_size, void* d_ws, size_t ws_size,
                              hipStream_t stream)
{
    const float*     grid  = (const float*)d_in[0];
    const float*     mesh  = (const float*)d_in[1];
    const float*     halo  = (const float*)d_in[2];
    const float*     efeat = (const float*)d_in[3];
    const long long* eidx  = (const long long*)d_in[4];
    const float *e_w1 = (const float*)d_in[5],  *e_b1 = (const float*)d_in[6],
                *e_w2 = (const float*)d_in[7],  *e_b2 = (const float*)d_in[8],
                *e_g  = (const float*)d_in[9],  *e_be = (const float*)d_in[10];
    const float *n_w1 = (const float*)d_in[11], *n_b1 = (const float*)d_in[12],
                *n_w2 = (const float*)d_in[13], *n_b2 = (const float*)d_in[14],
                *n_g  = (const float*)d_in[15], *n_be = (const float*)d_in[16];
    const float *g_w1 = (const float*)d_in[17], *g_b1 = (const float*)d_in[18],
                *g_w2 = (const float*)d_in[19], *g_b2 = (const float*)d_in[20],
                *g_g  = (const float*)d_in[21], *g_be = (const float*)d_in[22];

    float* out_grid = (float*)d_out;
    float* out_mesh = out_grid + (size_t)NGRID * HD;

    // ---- workspace layout ----
    unsigned char* ws = (unsigned char*)d_ws;
    size_t off = 0;
    float* agg = (float*)(ws + off);          off += (size_t)NMESH * HD * sizeof(float);
    __bf16* e_w1t = (__bf16*)(ws + off);      off += (size_t)512 * 1536 * 2;
    __bf16* e_w2t = (__bf16*)(ws + off);      off += (size_t)512 * 512  * 2;
    __bf16* n_w1t = (__bf16*)(ws + off);      off += (size_t)512 * 1024 * 2;
    __bf16* n_w2t = (__bf16*)(ws + off);      off += (size_t)512 * 512  * 2;
    __bf16* g_w1t = (__bf16*)(ws + off);      off += (size_t)512 * 512  * 2;
    __bf16* g_w2t = (__bf16*)(ws + off);      off += (size_t)512 * 512  * 2;

    // ---- zero aggregation buffer (required every call) ----
    const int nAgg = NMESH * HD;
    zero_kernel<<<(nAgg + 255) / 256, 256, 0, stream>>>(agg, nAgg);

    // ---- convert weights to transposed bf16 ----
    convert_transpose_kernel<<<(1536*512 + 255)/256, 256, 0, stream>>>(e_w1, e_w1t, 1536, 512);
    convert_transpose_kernel<<<( 512*512 + 255)/256, 256, 0, stream>>>(e_w2, e_w2t,  512, 512);
    convert_transpose_kernel<<<(1024*512 + 255)/256, 256, 0, stream>>>(n_w1, n_w1t, 1024, 512);
    convert_transpose_kernel<<<( 512*512 + 255)/256, 256, 0, stream>>>(n_w2, n_w2t,  512, 512);
    convert_transpose_kernel<<<( 512*512 + 255)/256, 256, 0, stream>>>(g_w1, g_w1t,  512, 512);
    convert_transpose_kernel<<<( 512*512 + 255)/256, 256, 0, stream>>>(g_w2, g_w2t,  512, 512);

    // ---- edge MLP + scatter-add into agg ----
    edge_mlp_kernel<<<E_EDGES / ROWS, 256, 0, stream>>>(
        efeat, mesh, halo, eidx, e_w1t, e_b1, e_w2t, e_b2, e_g, e_be, agg);

    // ---- node MLP (consumes agg) ----
    node_mlp_kernel<<<(NMESH + ROWS - 1) / ROWS, 256, 0, stream>>>(
        mesh, agg, n_w1t, n_b1, n_w2t, n_b2, n_g, n_be, out_mesh);

    // ---- grid MLP (independent) ----
    grid_mlp_kernel<<<NGRID / ROWS, 256, 0, stream>>>(
        grid, g_w1t, g_b1, g_w2t, g_b2, g_g, g_be, out_grid);
}
